// QuantizedE8P12Linear_Optimized_44899588112786
// MI455X (gfx1250) — compile-verified, matrix operand order checked by reading, not run on hardware
//
#include <hip/hip_runtime.h>
#include <hip/hip_bf16.h>

typedef __attribute__((ext_vector_type(16))) __bf16 v16bf;
typedef __attribute__((ext_vector_type(8)))  __bf16 v8bf;
typedef __attribute__((ext_vector_type(8)))  float  v8f;

#define N_DIM 4096   // input feature dim (= SU length)
#define M_DIM 4096   // output feature dim (= SV length)
#define ROWS  4096   // B*S = 2*2048

#define BM 128
#define BN 128
#define BK 64
#define LDT 88       // padded LDS K-stride (elements): 176B rows, conflict-free b128 loads
#define KITERS (N_DIM / BK)   // 64

// ---------------- Kernel 1: A = bf16( fwht(input * SU) / 64 ) ----------------
__global__ __launch_bounds__(256) void k_fwht_su(const float* __restrict__ x,
                                                 const float* __restrict__ SU,
                                                 __bf16* __restrict__ out)
{
    __shared__ float s[N_DIM];
    const int row = blockIdx.x;
    const int tid = threadIdx.x;
    const float* xr = x + (size_t)row * N_DIM;
    for (int i = tid; i < N_DIM; i += 256) s[i] = xr[i] * SU[i];
    __syncthreads();
    for (int h = 1; h < N_DIM; h <<= 1) {
        for (int t = tid; t < N_DIM / 2; t += 256) {
            int i = ((t & ~(h - 1)) << 1) | (t & (h - 1));
            float a = s[i], b = s[i + h];
            s[i] = a + b;
            s[i + h] = a - b;
        }
        __syncthreads();
    }
    __bf16* orow = out + (size_t)row * N_DIM;
    for (int i = tid; i < N_DIM; i += 256)
        orow[i] = (__bf16)(s[i] * 0.015625f);   // 1/sqrt(4096)
}

// ---------------- Kernel 2: W[j][k] = bf16(grid[Qidxs[j][k/8]][k%8]) ----------------
__global__ __launch_bounds__(256) void k_dequant(const float* __restrict__ grid,
                                                 const int* __restrict__ Qidxs,
                                                 __bf16* __restrict__ W)
{
    const int t = blockIdx.x * 256 + threadIdx.x;     // 0 .. M_DIM*(N_DIM/8)-1
    const int idx = Qidxs[t];
    const float4* g = (const float4*)(grid + (size_t)idx * 8);
    float4 a = g[0], b = g[1];
    v8bf o;
    o[0] = (__bf16)a.x; o[1] = (__bf16)a.y; o[2] = (__bf16)a.z; o[3] = (__bf16)a.w;
    o[4] = (__bf16)b.x; o[5] = (__bf16)b.y; o[6] = (__bf16)b.z; o[7] = (__bf16)b.w;
    *(v8bf*)(W + (size_t)t * 8) = o;                  // row-major: W[j*4096 + kb*8 ..]
}

// ---------------- Kernel 3: C[r][j] = sum_k A[r][k] * W[j][k]  (bf16 WMMA) ----------------
// Double-buffered LDS fed by async global->LDS DMA (ASYNCcnt), overlapped with WMMA.
__global__ __launch_bounds__(256) void k_gemm(const __bf16* __restrict__ A,
                                              const __bf16* __restrict__ W,
                                              float* __restrict__ C)
{
    __shared__ __bf16 As[2][BM * LDT];
    __shared__ __bf16 Bs[2][BN * LDT];

    const int tid  = threadIdx.x;
    const int lane = tid & 31;
    const int wave = tid >> 5;       // 0..7
    const int wm   = wave >> 2;      // 0..1  -> 64-row slice of M
    const int wn   = wave & 3;       // 0..3  -> 32-col slice of N
    const int row0 = blockIdx.y * BM;
    const int col0 = blockIdx.x * BN;

    const int r    = lane & 15;            // fragment row
    const int khal = (lane >> 4) * 8;      // lanes 0-15: K 0-7/16-23 ; 16-31: K 8-15/24-31

    // per-thread staging chunks: 4 x 16B for A-tile, 4 x 16B for B-tile
    const int ctr = tid >> 3;              // tile row       (0..31, +32*i)
    const int cko = (tid & 7) * 8;         // K offset (elements) of 16B chunk

    v8f acc[4][2];
#pragma unroll
    for (int mt = 0; mt < 4; ++mt)
#pragma unroll
        for (int nt = 0; nt < 2; ++nt) acc[mt][nt] = (v8f)(0.0f);

    // ---- async issue of one K-tile into LDS buffer `buf` (8 loads/thread) ----
    auto issue_tile = [&](int buf, int kt) {
#pragma unroll
        for (int i = 0; i < 4; ++i) {
            int tr = ctr + 32 * i;
            unsigned lA = (unsigned)(uintptr_t)&As[buf][tr * LDT + cko];
            unsigned lB = (unsigned)(uintptr_t)&Bs[buf][tr * LDT + cko];
            unsigned long long gA =
                (unsigned long long)&A[(size_t)(row0 + tr) * N_DIM + kt + cko];
            unsigned long long gB =
                (unsigned long long)&W[(size_t)(col0 + tr) * N_DIM + kt + cko];
            asm volatile("global_load_async_to_lds_b128 %0, %1, off"
                         :: "v"(lA), "v"(gA) : "memory");
            asm volatile("global_load_async_to_lds_b128 %0, %1, off"
                         :: "v"(lB), "v"(gB) : "memory");
            // speculative L2 prefetch two tiles ahead
            __builtin_prefetch(&A[(size_t)(row0 + tr) * N_DIM + kt + 2 * BK + cko], 0, 1);
            __builtin_prefetch(&W[(size_t)(col0 + tr) * N_DIM + kt + 2 * BK + cko], 0, 1);
        }
    };

    issue_tile(0, 0);    // prologue

    for (int it = 0; it < KITERS; ++it) {
        const int cur = it & 1;
        if (it + 1 < KITERS) {
            issue_tile(cur ^ 1, (it + 1) * BK);   // DMA next tile while we compute
            // async loads complete in order: <=8 outstanding => current tile landed
            asm volatile("s_wait_asynccnt 0x8" ::: "memory");
        } else {
            asm volatile("s_wait_asynccnt 0x0" ::: "memory");
        }
        __syncthreads();   // all waves' chunks of `cur` visible

        const __bf16* __restrict__ Ab = As[cur];
        const __bf16* __restrict__ Bb = Bs[cur];
#pragma unroll
        for (int ks = 0; ks < BK; ks += 32) {
            v16bf bfrag[2];
#pragma unroll
            for (int nt = 0; nt < 2; ++nt) {
                const __bf16* p = &Bb[(wn * 32 + nt * 16 + r) * LDT + ks + khal];
                v8bf lo = *(const v8bf*)p;          // K 0-7 (or 8-15)
                v8bf hi = *(const v8bf*)(p + 16);   // K 16-23 (or 24-31)
                bfrag[nt] = __builtin_shufflevector(lo, hi,
                    0, 1, 2, 3, 4, 5, 6, 7, 8, 9, 10, 11, 12, 13, 14, 15);
            }
#pragma unroll
            for (int mt = 0; mt < 4; ++mt) {
                const __bf16* p = &Ab[(wm * 64 + mt * 16 + r) * LDT + ks + khal];
                v8bf lo = *(const v8bf*)p;
                v8bf hi = *(const v8bf*)(p + 16);
                v16bf afrag = __builtin_shufflevector(lo, hi,
                    0, 1, 2, 3, 4, 5, 6, 7, 8, 9, 10, 11, 12, 13, 14, 15);
#pragma unroll
                for (int nt = 0; nt < 2; ++nt) {
                    acc[mt][nt] = __builtin_amdgcn_wmma_f32_16x16x32_bf16(
                        /*neg_a=*/false, afrag, /*neg_b=*/false, bfrag[nt],
                        /*c_mod=*/(short)0, acc[mt][nt],
                        /*reuse_a=*/false, /*reuse_b=*/false);
                }
            }
        }
        __syncthreads();   // everyone done reading `cur` before it is re-filled
    }

    // C/D layout: lane 0-15 -> N=lane, M=g ; lane 16-31 -> N=lane-16, M=g+8
    const int mh = (lane >> 4) * 8;
#pragma unroll
    for (int mt = 0; mt < 4; ++mt)
#pragma unroll
        for (int nt = 0; nt < 2; ++nt)
#pragma unroll
            for (int g = 0; g < 8; ++g) {
                int mrow = row0 + wm * 64 + mt * 16 + g + mh;
                int ncol = col0 + wn * 32 + nt * 16 + r;
                C[(size_t)mrow * M_DIM + ncol] = acc[mt][nt][g];
            }
}

// ---------------- Kernel 4: out = fwht(out)/64 * SV   (in place) ----------------
__global__ __launch_bounds__(256) void k_fwht_sv(float* __restrict__ y,
                                                 const float* __restrict__ SV)
{
    __shared__ float s[M_DIM];
    const int row = blockIdx.x;
    const int tid = threadIdx.x;
    float* yr = y + (size_t)row * M_DIM;
    for (int i = tid; i < M_DIM; i += 256) s[i] = yr[i];
    __syncthreads();
    for (int h = 1; h < M_DIM; h <<= 1) {
        for (int t = tid; t < M_DIM / 2; t += 256) {
            int i = ((t & ~(h - 1)) << 1) | (t & (h - 1));
            float a = s[i], b = s[i + h];
            s[i] = a + b;
            s[i + h] = a - b;
        }
        __syncthreads();
    }
    for (int i = tid; i < M_DIM; i += 256)
        yr[i] = s[i] * 0.015625f * SV[i];
}

extern "C" void kernel_launch(void* const* d_in, const int* in_sizes, int n_in,
                              void* d_out, int out_size, void* d_ws, size_t ws_size,
                              hipStream_t stream)
{
    const float* input = (const float*)d_in[0];   // (2,2048,4096) f32
    const float* SU    = (const float*)d_in[1];   // (4096,)       f32
    const float* SV    = (const float*)d_in[2];   // (4096,)       f32
    const float* grid  = (const float*)d_in[3];   // (65536,8)     f32
    const int*   Qidxs = (const int*)  d_in[4];   // (4096,512)    i32
    float*       out   = (float*)d_out;           // (2,2048,4096) f32

    // workspace: [0,32MB) bf16 A (rotated activations), [32MB,64MB) bf16 W
    __bf16* Abf = (__bf16*)d_ws;
    __bf16* Wbf = (__bf16*)((char*)d_ws + (size_t)ROWS * N_DIM * sizeof(__bf16));

    k_fwht_su<<<ROWS, 256, 0, stream>>>(input, SU, Abf);
    k_dequant<<<(M_DIM * (N_DIM / 8)) / 256, 256, 0, stream>>>(grid, Qidxs, Wbf);
    dim3 g(M_DIM / BN, ROWS / BM);
    k_gemm<<<g, 256, 0, stream>>>(Abf, Wbf, out);
    k_fwht_sv<<<ROWS, 256, 0, stream>>>(out, SV);
}
